// RelationNet_4312147165447
// MI455X (gfx1250) — matrix operand order repaired and checked by reading, not compiled, not run on hardware
//
#include <hip/hip_runtime.h>
#include <hip/hip_bf16.h>

typedef __attribute__((ext_vector_type(16))) _Float16 v16h;
typedef __attribute__((ext_vector_type(16))) __bf16   v16bf;
typedef __attribute__((ext_vector_type(8)))  float    v8f;

#define NB    16
#define CINC  3
#define IMG   64
#define SF    24
#define H1    32
#define H2    16
#define MODC  128
#define RNNC  128
#define FC1N  1024
#define NAC   2
#define DD    26      // SF + 2
#define K2    216     // SF*9 (im2col K for conv2)
#define K2P   224     // padded to 7*32
#define SIN_ZOFF (SF * 3 * 34)   // zero-pad region offset in sIn

// A-operand K index for 16-bit 16x32 A tiles (lane half `hi`, element idx 0..15)
__device__ constexpr int a_kidx(int idx, int hi) {
    return hi * 16 + idx;
}
// compile-time im2col base offset in sIn for a given K index (valid kk < K2)
__device__ constexpr int sin_base(int kk) {
    return ((kk / 9) * 3 + (kk % 9) / 3) * 34 + ((kk % 9) % 3);
}

// ---------------------------------------------------------------------------
// Kernel 1: conv1 (3x3 stride2 pad1) + bn + relu  -> f1 (f16)
// ---------------------------------------------------------------------------
__global__ void k_conv1(const float* __restrict__ img, const float* __restrict__ w,
                        const float* __restrict__ b, const float* __restrict__ g,
                        const float* __restrict__ be, const float* __restrict__ mu,
                        const float* __restrict__ var, _Float16* __restrict__ f1) {
    int idx = blockIdx.x * blockDim.x + threadIdx.x;          // NB*SF*H1*H1
    int ox = idx & 31, oy = (idx >> 5) & 31;
    int co = (idx >> 10) % SF, n = idx / (SF * 1024);
    float acc = b[co];
    #pragma unroll
    for (int ci = 0; ci < CINC; ++ci)
        #pragma unroll
        for (int ky = 0; ky < 3; ++ky) {
            int iy = 2 * oy - 1 + ky;
            if (iy < 0 || iy >= IMG) continue;
            #pragma unroll
            for (int kx = 0; kx < 3; ++kx) {
                int ix = 2 * ox - 1 + kx;
                if (ix < 0 || ix >= IMG) continue;
                acc += img[((n * CINC + ci) * IMG + iy) * IMG + ix] *
                       w[((co * CINC + ci) * 3 + ky) * 3 + kx];
            }
        }
    float inv = g[co] * rsqrtf(var[co] + 1e-5f);
    float y = (acc - mu[co]) * inv + be[co];
    f1[idx] = (_Float16)fmaxf(y, 0.f);
}

// ---------------------------------------------------------------------------
// Kernel 2: conv2 via im2col WMMA (f16 in, f32 acc) + bn + relu -> f2 (f32)
// One wave per (n, oy) tile: 16 rows (ox), K=216 (pad 224), N=24 (pad 32).
// A-operand addressing fully constant-folded; padding K reads a zeroed LDS
// region so the inner loop is branch-free (no exec masking).
// ---------------------------------------------------------------------------
__global__ void k_conv2_wmma(const _Float16* __restrict__ f1,
                             const float* __restrict__ w, const float* __restrict__ b,
                             const float* __restrict__ g, const float* __restrict__ be,
                             const float* __restrict__ mu, const float* __restrict__ var,
                             float* __restrict__ f2) {
    __shared__ _Float16 sB[K2P * 32];             // weights: [kk][co], zero padded
    __shared__ _Float16 sIn[SF * 3 * 34 + 34];    // input patch + zero pad

    int tile = blockIdx.x;                        // 256 tiles = (n_img, oy)
    int n_img = tile >> 4, oy = tile & 15;
    int l = threadIdx.x, n_ = l & 15, hi = l >> 4;

    for (int t = l; t < K2P * 32; t += 32) {
        int kk = t >> 5, co = t & 31;
        sB[t] = (kk < K2 && co < SF) ? (_Float16)w[co * K2 + kk] : (_Float16)0.f;
    }
    for (int t = l; t < SF * 3 * 34; t += 32) {
        int c = t % 34, r = (t / 34) % 3, ci = t / 102;
        int iy = 2 * oy - 1 + r, ix = c - 1;
        _Float16 v = (_Float16)0.f;
        if (iy >= 0 && iy < H1 && ix >= 0 && ix < H1)
            v = f1[((n_img * SF + ci) * H1 + iy) * H1 + ix];
        sIn[t] = v;
    }
    for (int t = l; t < 34; t += 32) sIn[SIN_ZOFF + t] = (_Float16)0.f;
    __syncthreads();

    v8f c0 = {}, c1 = {};
    #pragma unroll
    for (int ch = 0; ch < 7; ++ch) {
        v16h a;
        #pragma unroll
        for (int idx = 0; idx < 16; ++idx) {
            const int kk0 = ch * 32 + a_kidx(idx, 0);
            const int kk1 = ch * 32 + a_kidx(idx, 1);
            const int o0 = (kk0 < K2) ? sin_base(kk0) : SIN_ZOFF;
            const int o1 = (kk1 < K2) ? sin_base(kk1) : SIN_ZOFF;
            a[idx] = sIn[(hi ? o1 : o0) + 2 * n_];     // row m = ox = n_
        }
        v16h b0, b1;
        #pragma unroll
        for (int idx = 0; idx < 16; ++idx) {
            int kk = ch * 32 + hi * 16 + idx;
            b0[idx] = sB[kk * 32 + n_];
            b1[idx] = sB[kk * 32 + 16 + n_];
        }
        c0 = __builtin_amdgcn_wmma_f32_16x16x32_f16(false, a, false, b0, (short)0, c0, false, false);
        c1 = __builtin_amdgcn_wmma_f32_16x16x32_f16(false, a, false, b1, (short)0, c1, false, false);
    }

    int co0 = n_, co1 = 16 + n_;
    float inv0 = g[co0] * rsqrtf(var[co0] + 1e-5f);
    float bb0 = b[co0], m0 = mu[co0], e0 = be[co0];
    float inv1 = 0.f, bb1 = 0.f, m1 = 0.f, e1 = 0.f;
    if (co1 < SF) { inv1 = g[co1] * rsqrtf(var[co1] + 1e-5f); bb1 = b[co1]; m1 = mu[co1]; e1 = be[co1]; }
    #pragma unroll
    for (int r = 0; r < 8; ++r) {
        int ox = r + 8 * hi;
        float y0 = fmaxf((c0[r] + bb0 - m0) * inv0 + e0, 0.f);
        f2[((n_img * SF + co0) * H2 + oy) * H2 + ox] = y0;
        if (co1 < SF) {
            float y1 = fmaxf((c1[r] + bb1 - m1) * inv1 + e1, 0.f);
            f2[((n_img * SF + co1) * H2 + oy) * H2 + ox] = y1;
        }
    }
}

// ---------------------------------------------------------------------------
// Kernel 3: sum over HW -> sum_fc (16 x 26); coords sum = 1920 each
// ---------------------------------------------------------------------------
__global__ void k_sumfc(const float* __restrict__ f2, float* __restrict__ sum_fc) {
    int t = threadIdx.x;
    if (t < NB * SF) {
        int n = t / SF, c = t % SF;
        const float* p = f2 + (n * SF + c) * 256;
        float s = 0.f;
        for (int i = 0; i < 256; ++i) s += p[i];
        sum_fc[n * DD + c] = s;
    }
    if (t < 32) {  // coords: sum of meshgrid(0..15) over 16x16 grid = 1920
        int n = t >> 1, c = SF + (t & 1);
        sum_fc[n * DD + c] = 1920.f;
    }
}

// ---------------------------------------------------------------------------
// Kernel 4: fused head (one workgroup, 32 waves).
//  relations (f16 WMMA, f32 combine) -> fc1 (bf16 WMMA) -> fc2 (dot)
// ---------------------------------------------------------------------------
__global__ void k_head(const float* __restrict__ sum_fc, const float* __restrict__ ques,
                       const float* __restrict__ rel_w, const float* __restrict__ rel_b,
                       const float* __restrict__ fc1_w, const float* __restrict__ fc1_b,
                       const float* __restrict__ fc2_w, const float* __restrict__ fc2_b,
                       float* __restrict__ out) {
    __shared__ _Float16 sA1[NB * 32];       // sum_fc, K padded 26->32
    __shared__ _Float16 sA2[NB * RNNC];     // ques
    __shared__ float    relLds[NB * MODC];  // relations, f32 (bf16 path needs range)
    __shared__ _Float16 hLds[NB * FC1N];    // h = relu(fc1)

    int t = threadIdx.x;
    int wave = t >> 5, l = t & 31, n_ = l & 15, hi = l >> 4;

    for (int i = t; i < NB * 32; i += blockDim.x) {
        int m = i >> 5, k = i & 31;
        sA1[i] = (k < DD) ? (_Float16)sum_fc[m * DD + k] : (_Float16)0.f;
    }
    for (int i = t; i < NB * RNNC; i += blockDim.x)
        sA2[i] = (_Float16)ques[i];
    __syncthreads();

    // ---- relations: 8 N-tiles of 16, waves 0..7 ----
    if (wave < 8) {
        int nc = wave * 16 + n_;
        // part 1: sum_fc @ (Wi+Wj)^T, K = 26 -> one 32-chunk
        v16h a, b;
        #pragma unroll
        for (int idx = 0; idx < 16; ++idx)
            a[idx] = sA1[n_ * 32 + a_kidx(idx, hi)];
        #pragma unroll
        for (int idx = 0; idx < 16; ++idx) {
            int k = hi * 16 + idx;
            // indices stay inside the 180-wide row; mask value, not the load
            float wv = rel_w[nc * 180 + k] + rel_w[nc * 180 + DD + k];
            b[idx] = (_Float16)((k < DD) ? wv : 0.f);
        }
        v8f acc1 = {};
        acc1 = __builtin_amdgcn_wmma_f32_16x16x32_f16(false, a, false, b, (short)0, acc1, false, false);
        // part 2: ques @ Wq^T, K = 128 -> 4 chunks
        v8f acc2 = {};
        #pragma unroll
        for (int ch = 0; ch < 4; ++ch) {
            v16h aq, bq;
            #pragma unroll
            for (int idx = 0; idx < 16; ++idx)
                aq[idx] = sA2[n_ * RNNC + ch * 32 + a_kidx(idx, hi)];
            #pragma unroll
            for (int idx = 0; idx < 16; ++idx)
                bq[idx] = (_Float16)rel_w[nc * 180 + 2 * DD + ch * 32 + hi * 16 + idx];
            acc2 = __builtin_amdgcn_wmma_f32_16x16x32_f16(false, aq, false, bq, (short)0, acc2, false, false);
        }
        float rb = rel_b[nc];
        #pragma unroll
        for (int r = 0; r < 8; ++r) {
            int row = r + 8 * hi;
            relLds[row * MODC + nc] = 256.f * acc1[r] + 65536.f * (acc2[r] + rb);
        }
    }
    __syncthreads();

    // ---- h = relu(relations @ fc1_w^T + b): 64 N-tiles, 2 per wave, bf16 WMMA ----
    #pragma unroll
    for (int tt = 0; tt < 2; ++tt) {
        int tileN = wave * 2 + tt;
        int nc = tileN * 16 + n_;
        v8f acc = {};
        #pragma unroll
        for (int ch = 0; ch < 4; ++ch) {
            v16bf a, b;
            #pragma unroll
            for (int idx = 0; idx < 16; ++idx)
                a[idx] = (__bf16)relLds[n_ * MODC + ch * 32 + a_kidx(idx, hi)];
            #pragma unroll
            for (int idx = 0; idx < 16; ++idx)
                b[idx] = (__bf16)fc1_w[nc * MODC + ch * 32 + hi * 16 + idx];
            acc = __builtin_amdgcn_wmma_f32_16x16x32_bf16(false, a, false, b, (short)0, acc, false, false);
        }
        float fb = fc1_b[nc];
        #pragma unroll
        for (int r = 0; r < 8; ++r) {
            int row = r + 8 * hi;
            hLds[row * FC1N + nc] = (_Float16)fmaxf(acc[r] + fb, 0.f);
        }
    }
    __syncthreads();

    // ---- out = h @ fc2_w^T + b : 16x2, 32 threads ----
    if (t < NB * NAC) {
        int m = t >> 1, aidx = t & 1;
        float acc = fc2_b[aidx];
        for (int k = 0; k < FC1N; ++k)
            acc += (float)hLds[m * FC1N + k] * fc2_w[aidx * FC1N + k];
        out[m * NAC + aidx] = acc;
    }
}

// ---------------------------------------------------------------------------
extern "C" void kernel_launch(void* const* d_in, const int* in_sizes, int n_in,
                              void* d_out, int out_size, void* d_ws, size_t ws_size,
                              hipStream_t stream) {
    (void)in_sizes; (void)n_in; (void)out_size; (void)ws_size;
    const float* image   = (const float*)d_in[0];
    const float* ques    = (const float*)d_in[1];
    const float* conv1_w = (const float*)d_in[2];
    const float* conv1_b = (const float*)d_in[3];
    const float* bn1_g   = (const float*)d_in[4];
    const float* bn1_b   = (const float*)d_in[5];
    const float* bn1_m   = (const float*)d_in[6];
    const float* bn1_v   = (const float*)d_in[7];
    const float* conv2_w = (const float*)d_in[8];
    const float* conv2_b = (const float*)d_in[9];
    const float* bn2_g   = (const float*)d_in[10];
    const float* bn2_b   = (const float*)d_in[11];
    const float* bn2_m   = (const float*)d_in[12];
    const float* bn2_v   = (const float*)d_in[13];
    const float* rel_w   = (const float*)d_in[14];
    const float* rel_b   = (const float*)d_in[15];
    const float* fc1_w   = (const float*)d_in[16];
    const float* fc1_b   = (const float*)d_in[17];
    const float* fc2_w   = (const float*)d_in[18];
    const float* fc2_b   = (const float*)d_in[19];
    float* out = (float*)d_out;

    char* ws = (char*)d_ws;
    _Float16* f1  = (_Float16*)ws;                              // 16*24*32*32 f16 = 768 KiB
    float*    f2  = (float*)(ws + 786432);                      // 16*24*16*16 f32 = 384 KiB
    float*    sfc = (float*)(ws + 786432 + 393216);             // 16*26 f32

    k_conv1<<<1536, 256, 0, stream>>>(image, conv1_w, conv1_b, bn1_g, bn1_b, bn1_m, bn1_v, f1);
    k_conv2_wmma<<<256, 32, 0, stream>>>(f1, conv2_w, conv2_b, bn2_g, bn2_b, bn2_m, bn2_v, f2);
    k_sumfc<<<1, 384, 0, stream>>>(f2, sfc);
    k_head<<<1, 1024, 0, stream>>>(sfc, ques, rel_w, rel_b, fc1_w, fc1_b, fc2_w, fc2_b, out);
}
